// MultiHeadAttention_46840913330573
// MI455X (gfx1250) — compile-verified, matrix operand order checked by reading, not run on hardware
//
#include <hip/hip_runtime.h>
#include <hip/hip_bf16.h>

// ---------- types ----------
typedef __attribute__((ext_vector_type(16))) __bf16          v16bf;
typedef __attribute__((ext_vector_type(8)))  float           v8f;
typedef __attribute__((ext_vector_type(16))) unsigned short  v16u;
typedef __attribute__((ext_vector_type(4)))  int             v4i;
using u16 = unsigned short;

typedef __attribute__((address_space(1))) v4i gv4i;   // global (AS1) int4
typedef __attribute__((address_space(3))) v4i lv4i;   // LDS (AS3) int4

#define NCTX   2048
#define BATCH  2
#define DMODEL 1024
#define HEADS  16
#define DK     64
#define MROWS  (BATCH * NCTX)   /* 4096 */
#define SCALE  0.125f           /* 1/sqrt(64) */

// ---------- async global->LDS staging (gfx1250 GLOBAL_LOAD_ASYNC_TO_LDS_B128) ----------
#if defined(__has_builtin)
#if __has_builtin(__builtin_amdgcn_global_load_async_to_lds_b128)
#define HAVE_ASYNC_LDS 1
#endif
#if __has_builtin(__builtin_amdgcn_s_wait_asynccnt)
#define HAVE_WAIT_ASYNC 1
#endif
#endif

__device__ __forceinline__ void async_copy16(const u16* g, u16* l) {
#if defined(HAVE_ASYNC_LDS)
  __builtin_amdgcn_global_load_async_to_lds_b128((gv4i*)g, (lv4i*)l, 0, 0);
#else
  *reinterpret_cast<uint4*>(l) = *reinterpret_cast<const uint4*>(g);
#endif
}

__device__ __forceinline__ void async_wait() {
#if defined(HAVE_ASYNC_LDS)
#if defined(HAVE_WAIT_ASYNC)
  __builtin_amdgcn_s_wait_asynccnt(0);
#else
  asm volatile("s_wait_asynccnt 0x0" ::: "memory");
#endif
#endif
}

// ---------- helpers ----------
__device__ __forceinline__ u16 f2bf(float f) {
  unsigned u = __float_as_uint(f);
  u += 0x7FFFu + ((u >> 16) & 1u);           // round-to-nearest-even
  return (u16)(u >> 16);
}

// Load a 16x32 bf16 WMMA fragment (A-layout; B-layout via swapped strides).
// Element e of lane L maps to (idx = L&15, k = (e&7) + ((e>>3)<<4) + ((L>>4)<<3))
// per CDNA5 ISA 7.12.2 (16-bit A-matrix 16x32). With k_stride==1 this lowers
// to 2 x b128 loads per lane.
__device__ __forceinline__ v16bf load_frag(const u16* p, int idx_stride, int k_stride) {
  const int lane = threadIdx.x & 31;
  const int hlf  = lane >> 4;
  const int idx  = lane & 15;
  v16u t;
#pragma unroll
  for (int e = 0; e < 16; ++e) {
    const int k = (e & 7) + ((e >> 3) << 4) + (hlf << 3);
    t[e] = p[idx * idx_stride + k * k_stride];
  }
  return __builtin_bit_cast(v16bf, t);
}

// ---------- fp32 -> bf16 conversion ----------
__global__ __launch_bounds__(256)
void cvt_f32_bf16(const float* __restrict__ in, u16* __restrict__ out, int count) {
  int i = blockIdx.x * 256 + threadIdx.x;
  if (i < count) out[i] = f2bf(in[i]);
}

// fp32 [K,N] -> bf16 transposed [N,K]  (weights, one-time)
__global__ __launch_bounds__(256)
void cvt_w_t(const float* __restrict__ in, u16* __restrict__ out) {
  int idx = blockIdx.x * 256 + threadIdx.x;       // over DMODEL*DMODEL
  int n = idx >> 10, k = idx & (DMODEL - 1);
  out[idx] = f2bf(in[(size_t)k * DMODEL + n]);
}

// ---------- generic bf16 GEMM: C[M,N] = A[M,K] @ B[K,N] + bias ----------
// B is supplied TRANSPOSED (BT[N,K]) so every fragment load is k-contiguous.
// Block tile 128x64, 8 waves, each wave 32x32 via 2x2 wmma_f32_16x16x32_bf16.
// MODE: 0 = bf16 row-major [M,N]; 1 = bf16 V-transposed [b,h,dk,n]; 2 = f32 row-major.
template <int MODE>
__global__ __launch_bounds__(256)
void gemm_bf16(const u16* __restrict__ A, const u16* __restrict__ BT,
               const float* __restrict__ bias, void* __restrict__ Cout,
               int N, int K) {
  __shared__ u16 As[128 * 40];   // 128 rows x (32 + 8 pad)
  __shared__ u16 Bs[64 * 40];    // 64 n-rows x (32 + 8 pad)

  const int tid  = threadIdx.x;
  const int m0   = blockIdx.y * 128;
  const int n0   = blockIdx.x * 64;
  const int wave = tid >> 5;
  const int wm   = (wave & 3) * 32;
  const int wn   = (wave >> 2) * 32;

  v8f acc[2][2] = {};

  for (int k0 = 0; k0 < K; k0 += 32) {
    // stage A tile (128 rows x 32 k)
#pragma unroll
    for (int it = 0; it < 2; ++it) {
      int slot = tid + it * 256;
      int row = slot >> 2, quad = slot & 3;
      async_copy16(&A[(size_t)(m0 + row) * K + k0 + quad * 8], &As[row * 40 + quad * 8]);
    }
    // stage BT tile (64 n-rows x 32 k)
    {
      int row = tid >> 2, quad = tid & 3;
      async_copy16(&BT[(size_t)(n0 + row) * K + k0 + quad * 8], &Bs[row * 40 + quad * 8]);
    }
    if (k0 + 32 < K) {
      __builtin_prefetch(&A[(size_t)(m0 + (tid >> 1)) * K + k0 + 32], 0, 1);
      __builtin_prefetch(&BT[(size_t)(n0 + (tid >> 2)) * K + k0 + 32], 0, 1);
    }
    async_wait();
    __syncthreads();

    v16bf a0 = load_frag(&As[(wm + 0)  * 40], 40, 1);
    v16bf a1 = load_frag(&As[(wm + 16) * 40], 40, 1);
    v16bf b0 = load_frag(&Bs[(wn + 0)  * 40], 40, 1);   // (k,n) = BT[n][k]
    v16bf b1 = load_frag(&Bs[(wn + 16) * 40], 40, 1);

    acc[0][0] = __builtin_amdgcn_wmma_f32_16x16x32_bf16(false, a0, false, b0, (short)0, acc[0][0], false, false);
    acc[0][1] = __builtin_amdgcn_wmma_f32_16x16x32_bf16(false, a0, false, b1, (short)0, acc[0][1], false, false);
    acc[1][0] = __builtin_amdgcn_wmma_f32_16x16x32_bf16(false, a1, false, b0, (short)0, acc[1][0], false, false);
    acc[1][1] = __builtin_amdgcn_wmma_f32_16x16x32_bf16(false, a1, false, b1, (short)0, acc[1][1], false, false);
    __syncthreads();
  }

  const int lane = tid & 31, hlf = lane >> 4, nc = lane & 15;
#pragma unroll
  for (int ti = 0; ti < 2; ++ti)
#pragma unroll
    for (int tj = 0; tj < 2; ++tj)
#pragma unroll
      for (int r = 0; r < 8; ++r) {
        int m  = m0 + wm + ti * 16 + r + 8 * hlf;
        int nn = n0 + wn + tj * 16 + nc;
        float v = acc[ti][tj][r] + bias[nn];
        if constexpr (MODE == 0) {
          ((u16*)Cout)[(size_t)m * N + nn] = f2bf(v);
        } else if constexpr (MODE == 1) {
          // V transposed: [b, h, dk, n] for k-contiguous P@V fragments
          int bb = m >> 11, ii = m & (NCTX - 1);
          int hh = nn >> 6, dc = nn & (DK - 1);
          ((u16*)Cout)[((size_t)(bb * HEADS + hh) * DK + dc) * NCTX + ii] = f2bf(v);
        } else {
          ((float*)Cout)[(size_t)m * N + nn] = v;
        }
      }
}

// ---------- pass A: per-row softmax stats (online max/sum) ----------
__global__ __launch_bounds__(256)
void attn_stats(const u16* __restrict__ Qb, const u16* __restrict__ Kb,
                float* __restrict__ stats) {
  __shared__ u16 Ks[32 * 72];
  const int tid  = threadIdx.x;
  const int wave = tid >> 5, lane = tid & 31, hlf = lane >> 4, nc = lane & 15;
  const int bb = blockIdx.z, hh = blockIdx.y;
  const int i0 = blockIdx.x * 128 + wave * 16;

  const u16* qbase = Qb + (size_t)(bb * NCTX + i0) * DMODEL + hh * DK;
  v16bf qa0 = load_frag(qbase + 0,  DMODEL, 1);
  v16bf qa1 = load_frag(qbase + 32, DMODEL, 1);

  float mrow[8], lrow[8];
#pragma unroll
  for (int r = 0; r < 8; ++r) { mrow[r] = -1e30f; lrow[r] = 0.f; }

  for (int j0 = 0; j0 < NCTX; j0 += 32) {
    {
      int row = tid >> 3, quad = tid & 7;
      async_copy16(&Kb[(size_t)(bb * NCTX + j0 + row) * DMODEL + hh * DK + quad * 8],
                   &Ks[row * 72 + quad * 8]);
    }
    async_wait();
    __syncthreads();
#pragma unroll
    for (int t = 0; t < 2; ++t) {
      v8f s = {};
      v16bf kb0 = load_frag(&Ks[(t * 16) * 72 + 0],  72, 1);  // (k=d, n=j) = K[j][d]
      v16bf kb1 = load_frag(&Ks[(t * 16) * 72 + 32], 72, 1);
      s = __builtin_amdgcn_wmma_f32_16x16x32_bf16(false, qa0, false, kb0, (short)0, s, false, false);
      s = __builtin_amdgcn_wmma_f32_16x16x32_bf16(false, qa1, false, kb1, (short)0, s, false, false);
#pragma unroll
      for (int r = 0; r < 8; ++r) {
        float sv = s[r] * SCALE;
        float tm = sv;
#pragma unroll
        for (int msk = 1; msk < 16; msk <<= 1) tm = fmaxf(tm, __shfl_xor(tm, msk, 32));
        float nm = fmaxf(mrow[r], tm);
        float e  = __expf(sv - nm);
#pragma unroll
        for (int msk = 1; msk < 16; msk <<= 1) e += __shfl_xor(e, msk, 32);
        lrow[r] = lrow[r] * __expf(mrow[r] - nm) + e;
        mrow[r] = nm;
      }
    }
    __syncthreads();
  }
  if (nc == 0) {
#pragma unroll
    for (int r = 0; r < 8; ++r) {
      size_t row = (size_t)((bb * HEADS + hh) * NCTX) + i0 + r + 8 * hlf;
      stats[row * 2 + 0] = mrow[r];
      stats[row * 2 + 1] = lrow[r];
    }
  }
}

// ---------- pass B: attn output + P@V context ----------
// V supplied transposed: Vt[b, h, dk, n]
__global__ __launch_bounds__(256)
void attn_apply(const u16* __restrict__ Qb, const u16* __restrict__ Kb,
                const u16* __restrict__ Vt, const float* __restrict__ stats,
                float* __restrict__ attn, u16* __restrict__ ctx) {
  __shared__ u16 Ks[32 * 72];
  __shared__ u16 Vs[64 * 40];       // 64 dk-rows x (32 j + 8 pad)
  __shared__ u16 Ps[8 * 16 * 32];   // per-wave P staging (C-layout -> A-layout)

  const int tid  = threadIdx.x;
  const int wave = tid >> 5, lane = tid & 31, hlf = lane >> 4, nc = lane & 15;
  const int bb = blockIdx.z, hh = blockIdx.y;
  const int i0 = blockIdx.x * 128 + wave * 16;

  const u16* qbase = Qb + (size_t)(bb * NCTX + i0) * DMODEL + hh * DK;
  v16bf qa0 = load_frag(qbase + 0,  DMODEL, 1);
  v16bf qa1 = load_frag(qbase + 32, DMODEL, 1);

  float mrow[8], linv[8];
#pragma unroll
  for (int r = 0; r < 8; ++r) {
    size_t row = (size_t)((bb * HEADS + hh) * NCTX) + i0 + r + 8 * hlf;
    mrow[r] = stats[row * 2 + 0];
    linv[r] = 1.0f / stats[row * 2 + 1];
  }

  v8f o[4] = {};

  for (int j0 = 0; j0 < NCTX; j0 += 32) {
    {
      int row = tid >> 3, quad = tid & 7;
      async_copy16(&Kb[(size_t)(bb * NCTX + j0 + row) * DMODEL + hh * DK + quad * 8],
                   &Ks[row * 72 + quad * 8]);
    }
    {
      int dc = tid >> 2, quad = tid & 3;
      async_copy16(&Vt[((size_t)(bb * HEADS + hh) * DK + dc) * NCTX + j0 + quad * 8],
                   &Vs[dc * 40 + quad * 8]);
    }
    async_wait();
    __syncthreads();

#pragma unroll
    for (int t = 0; t < 2; ++t) {
      v8f s = {};
      v16bf kb0 = load_frag(&Ks[(t * 16) * 72 + 0],  72, 1);
      v16bf kb1 = load_frag(&Ks[(t * 16) * 72 + 32], 72, 1);
      s = __builtin_amdgcn_wmma_f32_16x16x32_bf16(false, qa0, false, kb0, (short)0, s, false, false);
      s = __builtin_amdgcn_wmma_f32_16x16x32_bf16(false, qa1, false, kb1, (short)0, s, false, false);
#pragma unroll
      for (int r = 0; r < 8; ++r) {
        float p = __expf(s[r] * SCALE - mrow[r]) * linv[r];
        // attn layout: [i, j, b, h]
        size_t ai = (((size_t)(i0 + r + 8 * hlf) * NCTX + (j0 + t * 16 + nc)) * BATCH + bb) * HEADS + hh;
        attn[ai] = p;
        Ps[wave * 512 + (r + 8 * hlf) * 32 + t * 16 + nc] = f2bf(p);
      }
    }
    __syncthreads();

    v16bf pa = load_frag(&Ps[wave * 512], 32, 1);           // A: 16 i-rows x 32 j
#pragma unroll
    for (int nt = 0; nt < 4; ++nt) {
      v16bf vb = load_frag(&Vs[(nt * 16) * 40], 40, 1);     // (k=j, n=dc) = Vt[dc][j]
      o[nt] = __builtin_amdgcn_wmma_f32_16x16x32_bf16(false, pa, false, vb, (short)0, o[nt], false, false);
    }
    __syncthreads();
  }

#pragma unroll
  for (int nt = 0; nt < 4; ++nt)
#pragma unroll
    for (int r = 0; r < 8; ++r) {
      size_t ci = (size_t)(bb * NCTX + i0 + r + 8 * hlf) * DMODEL + hh * DK + nt * 16 + nc;
      ctx[ci] = f2bf(o[nt][r]);
    }
}

// ---------- launch ----------
extern "C" void kernel_launch(void* const* d_in, const int* in_sizes, int n_in,
                              void* d_out, int out_size, void* d_ws, size_t ws_size,
                              hipStream_t stream) {
  (void)in_sizes; (void)n_in; (void)out_size; (void)ws_size;
  const float* x  = (const float*)d_in[0];
  const float* Wq = (const float*)d_in[1];
  const float* bq = (const float*)d_in[2];
  const float* Wk = (const float*)d_in[3];
  const float* bk = (const float*)d_in[4];
  const float* Wv = (const float*)d_in[5];
  const float* bv = (const float*)d_in[6];
  const float* Wo = (const float*)d_in[7];
  const float* bo = (const float*)d_in[8];

  const size_t SZ_X = (size_t)MROWS * DMODEL;     // 4,194,304 elems
  const size_t SZ_W = (size_t)DMODEL * DMODEL;    // 1,048,576 elems

  u16* xb   = (u16*)d_ws;
  u16* wqb  = xb  + SZ_X;      // all weights stored transposed bf16 [N,K]
  u16* wkb  = wqb + SZ_W;
  u16* wvb  = wkb + SZ_W;
  u16* wob  = wvb + SZ_W;
  u16* qb   = wob + SZ_W;      // [b*n, d] bf16
  u16* kb   = qb  + SZ_X;      // [b*n, d] bf16
  u16* vtb  = kb  + SZ_X;      // [b, h, dk, n] bf16
  u16* ctxb = vtb + SZ_X;      // [b*n, d] bf16
  float* stats = (float*)(ctxb + SZ_X);           // [b*h*n][2]

  cvt_f32_bf16<<<(unsigned)((SZ_X + 255) / 256), 256, 0, stream>>>(x, xb, (int)SZ_X);
  cvt_w_t<<<(unsigned)((SZ_W + 255) / 256), 256, 0, stream>>>(Wq, wqb);
  cvt_w_t<<<(unsigned)((SZ_W + 255) / 256), 256, 0, stream>>>(Wk, wkb);
  cvt_w_t<<<(unsigned)((SZ_W + 255) / 256), 256, 0, stream>>>(Wv, wvb);
  cvt_w_t<<<(unsigned)((SZ_W + 255) / 256), 256, 0, stream>>>(Wo, wob);

  dim3 gg(DMODEL / 64, MROWS / 128);
  gemm_bf16<0><<<gg, 256, 0, stream>>>(xb, wqb, bq, qb,  DMODEL, DMODEL);
  gemm_bf16<0><<<gg, 256, 0, stream>>>(xb, wkb, bk, kb,  DMODEL, DMODEL);
  gemm_bf16<1><<<gg, 256, 0, stream>>>(xb, wvb, bv, vtb, DMODEL, DMODEL);

  dim3 ga(NCTX / 128, HEADS, BATCH);
  attn_stats<<<ga, 256, 0, stream>>>(qb, kb, stats);

  float* out  = (float*)d_out;
  float* attn = out + (size_t)BATCH * NCTX * DMODEL;
  attn_apply<<<ga, 256, 0, stream>>>(qb, kb, vtb, stats, attn, ctxb);

  gemm_bf16<2><<<gg, 256, 0, stream>>>(ctxb, wob, bo, out, DMODEL, DMODEL);
}
